// NSF_CL_48198122995923
// MI455X (gfx1250) — matrix operand order repaired and checked by reading, not compiled, not run on hardware
//
#include <hip/hip_runtime.h>

// ---------------------------------------------------------------------------
// Fused neural-spline-flow coupling kernel for MI455X (gfx1250, wave32, WMMA)
//   - prep pass: weights -> slab-major transposed bf16 hi/lo in workspace
//   - main:      32 rows/WG, 256 threads (8 waves), GEMMs via
//                v_wmma_f32_16x16x32_bf16 with split-bf16 (hi+lo) accumulation
//   - weight slabs staged global->LDS via GLOBAL_LOAD_ASYNC_TO_LDS_B128
//     (ASYNCcnt) when the builtin exists; uint4 ds_store fallback otherwise
//   - all activations + RQS spline resident in LDS (~262KB of 320KB WGP LDS)
// ---------------------------------------------------------------------------

#define ROWS    32
#define THREADS 256
#define NWAVES  8
#define HID     256
#define OUTD    736          // (3K-1)*HALF = 23*32
#define KB      8
#define MIN_BW  0.001f
#define MIN_BH  0.001f
#define MIN_D   0.001f

typedef __attribute__((ext_vector_type(16))) __bf16 bf16x16;
typedef __attribute__((ext_vector_type(8)))  __bf16 bf16x8;
typedef __attribute__((ext_vector_type(8)))  float  f32x8;

// ---- LDS layout (bytes). Total 262272 < 320KB WGP LDS. All 16B aligned. ----
#define OFF_XLOW 0u        // 32*32*4   = 4096   f32 lower half
#define OFF_XUP  4096u     // 32*32*4   = 4096   f32 upper half
#define OFF_LD   8192u     // 32*4      = 128    per-row log-det accumulators
#define OFF_AH   8320u     // 32*256*2  = 16384  activation hi (bf16)
#define OFF_AL   24704u    // 32*256*2  = 16384  activation lo (bf16)
#define OFF_HH   41088u    // 32*256*2  = 16384  hidden hi
#define OFF_HL   57472u    // 32*256*2  = 16384  hidden lo
#define OFF_WB   73856u    // 736*128   = 94208  weight slab (hi block + lo block)
#define OFF_OUT  168064u   // 32*736*4  = 94208  f32 layer-2 output
#define SMEM_BYTES 262272u

// Async global->LDS copy path (gfx1250): guarded so absence falls back cleanly.
#if defined(__HIP_DEVICE_COMPILE__) && \
    __has_builtin(__builtin_amdgcn_global_load_async_to_lds_b128) && \
    __has_builtin(__builtin_amdgcn_s_wait_asynccnt)
#define USE_ASYNC 1
#define AS_GLOBAL __attribute__((address_space(1)))
#define AS_LOCAL  __attribute__((address_space(3)))
// Match the builtin's expected pointee exactly: gcc-style int4 vector.
typedef int v4i_b __attribute__((vector_size(16)));
#else
#define USE_ASYNC 0
#endif

__device__ __forceinline__ float fast_tanh(float x) {
#if defined(__HIP_DEVICE_COMPILE__) && __has_builtin(__builtin_amdgcn_tanhf)
    return __builtin_amdgcn_tanhf(x);      // v_tanh_f32 on gfx1250
#else
    return tanhf(x);
#endif
}

__device__ __forceinline__ float softplus_f(float x) {
    return fmaxf(x, 0.f) + log1pf(__expf(-fabsf(x)));
}

// A-fragment: 16x32 bf16, row-major LDS [row][k] (stride ASTR elements).
// ISA layout: lane<16 -> M=lane,   K 0..7 then 16..23
//             lane>=16 -> M=lane-16, K 8..15 then 24..31
__device__ __forceinline__ bf16x16 load_a_frag(const __bf16* base, int stride,
                                               int m, int kbase, int lane) {
    int row  = m * 16 + (lane & 15);
    int koff = (lane >> 4) << 3;
    const __bf16* p = base + row * stride + kbase + koff;
    bf16x8 a = *(const bf16x8*)p;          // K = koff .. koff+7
    bf16x8 b = *(const bf16x8*)(p + 16);   // K = 16+koff .. 16+koff+7
    return __builtin_shufflevector(a, b, 0,1,2,3,4,5,6,7,8,9,10,11,12,13,14,15);
}

// B-fragment: 32x16 bf16 from weight slab stored transposed [out][k], k stride 32.
// ISA layout: col = lane&15; lane<16 -> K 0..7 then 16..23, lane>=16 -> K 8..15 then 24..31
__device__ __forceinline__ bf16x16 load_b_frag(const __bf16* wb, int n, int lane) {
    int col = n * 16 + (lane & 15);
    const __bf16* p = wb + col * 32 + ((lane >> 4) << 3);
    bf16x8 a = *(const bf16x8*)p;
    bf16x8 b = *(const bf16x8*)(p + 16);
    return __builtin_shufflevector(a, b, 0,1,2,3,4,5,6,7,8,9,10,11,12,13,14,15);
}

// Contiguous slab copy global->LDS (nbytes is a multiple of 4096).
__device__ __forceinline__ void stage_slab(const __bf16* __restrict__ src,
                                           __bf16* dst, int nbytes, int tid) {
#if USE_ASYNC
    const char* gs = (const char*)src;
    char*       ld = (char*)dst;
    for (int off = tid * 16; off < nbytes; off += THREADS * 16) {
        __builtin_amdgcn_global_load_async_to_lds_b128(
            (AS_GLOBAL v4i_b*)(gs + off),
            (AS_LOCAL  v4i_b*)(ld + off), 0, 0);
    }
    __builtin_amdgcn_s_wait_asynccnt(0);   // my wave's async copies done
#else
    const uint4* s = (const uint4*)src;
    uint4*       d = (uint4*)dst;
    int n16 = nbytes >> 4;
    for (int i = tid; i < n16; i += THREADS) d[i] = s[i];
#endif
}

// One dense layer: D(32 x NOUT) = act(A(32 x KTOT) @ W + b)
// Wt: preprocessed slab-major weights; slab ks = NOUT*128 bytes
//     = [NOUT*32 bf16 hi][NOUT*32 bf16 lo], each [out][k] with k stride 32.
template <int KTOT, int NOUT, int ASTR, int TPW, bool DO_TANH>
__device__ void gemm_layer(const __bf16* AH, const __bf16* AL,
                           const __bf16* __restrict__ Wt,
                           const float* __restrict__ Bg,
                           __bf16* WB,
                           __bf16* DH, __bf16* DL, float* DF, int tid) {
    const int lane = tid & 31;
    const int wave = __builtin_amdgcn_readfirstlane(tid >> 5);  // SGPR -> scalar
                                                                // branches, EXEC
                                                                // untouched @WMMA
    const int NT = NOUT / 16;              // N tiles (16 or 46); M tiles = 2
    __bf16* WH = WB;
    __bf16* WL = WB + NOUT * 32;

    f32x8 acc[TPW];
#pragma unroll
    for (int i = 0; i < TPW; ++i)
        acc[i] = f32x8{0.f,0.f,0.f,0.f,0.f,0.f,0.f,0.f};

    for (int ks = 0; ks < KTOT / 32; ++ks) {
        __syncthreads();                   // Wbuf free of previous readers
        stage_slab(Wt + (size_t)ks * (NOUT * 64), WB, NOUT * 128, tid);
        __syncthreads();
#pragma unroll
        for (int i = 0; i < TPW; ++i) {
            int t = wave + i * NWAVES;     // scalar -> uniform branch
            if (t < 2 * NT) {
                int m = t / NT, n = t - m * NT;
                bf16x16 ah = load_a_frag(AH, ASTR, m, ks * 32, lane);
                bf16x16 al = load_a_frag(AL, ASTR, m, ks * 32, lane);
                bf16x16 bh = load_b_frag(WH, n, lane);
                bf16x16 bl = load_b_frag(WL, n, lane);
                acc[i] = __builtin_amdgcn_wmma_f32_16x16x32_bf16(false, ah, false, bh,
                             (short)0, acc[i], false, false);
                acc[i] = __builtin_amdgcn_wmma_f32_16x16x32_bf16(false, ah, false, bl,
                             (short)0, acc[i], false, false);
                acc[i] = __builtin_amdgcn_wmma_f32_16x16x32_bf16(false, al, false, bh,
                             (short)0, acc[i], false, false);
            }
        }
    }
    __syncthreads();   // all A/W reads done before epilogue writes

    // C/D layout: VGPR r, lane L -> row = m*16 + (L>=16 ? 8:0) + r, col = n*16+(L&15)
#pragma unroll
    for (int i = 0; i < TPW; ++i) {
        int t = wave + i * NWAVES;
        if (t < 2 * NT) {
            int m = t / NT, n = t - (t / NT) * NT;
            int col  = n * 16 + (lane & 15);
            float bias = Bg[col];
            int rbase = m * 16 + ((lane >> 4) << 3);
#pragma unroll
            for (int r = 0; r < 8; ++r) {
                float v = acc[i][r] + bias;
                int row = rbase + r;
                if (DO_TANH) {
                    v = fast_tanh(v);
                    __bf16 h = (__bf16)v;
                    DH[row * HID + col] = h;
                    DL[row * HID + col] = (__bf16)(v - (float)h);
                } else {
                    DF[row * NOUT + col] = v;
                }
            }
        }
    }
    __syncthreads();
}

// Rational-quadratic spline transform of tgt (32x32 f32 in LDS) using the
// 32x736 parameter block in LDS. One thread per 4 (row, j) pairs.
__device__ void rqs_block(const float* __restrict__ ob, float* __restrict__ tgt,
                          float* ldacc, int tid) {
    int row = tid >> 3;              // 0..31
    int j0  = (tid & 7) << 2;        // 0,4,...,28
    float ldsum = 0.f;
#pragma unroll 1
    for (int jj = 0; jj < 4; ++jj) {
        int j = j0 + jj;
        const float* p = ob + row * OUTD + j * 23;

        // ---- widths: uW = 2B*softmax(raw); widths = softmax(uW) ----
        float a[KB];
        float mx = -1e30f;
#pragma unroll
        for (int k = 0; k < KB; ++k) { a[k] = p[k]; mx = fmaxf(mx, a[k]); }
        float s = 0.f;
#pragma unroll
        for (int k = 0; k < KB; ++k) { a[k] = __expf(a[k] - mx); s += a[k]; }
        float sc = 6.0f / s;                       // 2B = 6
        float m2 = -1e30f;
#pragma unroll
        for (int k = 0; k < KB; ++k) { a[k] *= sc; m2 = fmaxf(m2, a[k]); }
        s = 0.f;
#pragma unroll
        for (int k = 0; k < KB; ++k) { a[k] = __expf(a[k] - m2); s += a[k]; }
        float invs = (1.0f - MIN_BW * KB) / s;
        float cw[KB + 1];
        cw[0] = -3.f;
        float c = 0.f;
#pragma unroll
        for (int k = 0; k < KB; ++k) { c += MIN_BW + a[k] * invs; cw[k + 1] = fmaf(6.f, c, -3.f); }
        cw[KB] = 3.f;

        // ---- heights (same pipeline on p[8..15]) ----
        float b[KB];
        mx = -1e30f;
#pragma unroll
        for (int k = 0; k < KB; ++k) { b[k] = p[KB + k]; mx = fmaxf(mx, b[k]); }
        s = 0.f;
#pragma unroll
        for (int k = 0; k < KB; ++k) { b[k] = __expf(b[k] - mx); s += b[k]; }
        sc = 6.0f / s;
        m2 = -1e30f;
#pragma unroll
        for (int k = 0; k < KB; ++k) { b[k] *= sc; m2 = fmaxf(m2, b[k]); }
        s = 0.f;
#pragma unroll
        for (int k = 0; k < KB; ++k) { b[k] = __expf(b[k] - m2); s += b[k]; }
        float invsh = (1.0f - MIN_BH * KB) / s;
        float ch[KB + 1];
        ch[0] = -3.f;
        c = 0.f;
#pragma unroll
        for (int k = 0; k < KB; ++k) { c += MIN_BH + b[k] * invsh; ch[k + 1] = fmaf(6.f, c, -3.f); }
        ch[KB] = 3.f;

        // ---- derivatives: dv = MIN_D + softplus(softplus(raw)); ends = 1 ----
        float dv[KB + 1];
        dv[0] = 1.f; dv[KB] = 1.f;
#pragma unroll
        for (int t = 0; t < KB - 1; ++t)
            dv[t + 1] = MIN_D + softplus_f(softplus_f(p[2 * KB + t]));

        // ---- bin search ----
        float xv = tgt[row * 32 + j];
        bool inside = (xv >= -3.f) && (xv <= 3.f);
        float xin = fminf(fmaxf(xv, -3.f), 3.f);
        int cnt = 0;
#pragma unroll
        for (int i = 0; i <= KB; ++i) {
            float loc = cw[i] + ((i == KB) ? 1e-6f : 0.f);
            cnt += (xin >= loc) ? 1 : 0;
        }
        int bin = cnt - 1;
        bin = bin < 0 ? 0 : (bin > KB - 1 ? KB - 1 : bin);

        float icw = 0.f, iw = 1.f, ich = 0.f, ih = 1.f, id0 = 1.f, id1 = 1.f;
#pragma unroll
        for (int i = 0; i < KB; ++i) {
            if (i == bin) {
                icw = cw[i]; iw = cw[i + 1] - cw[i];
                ich = ch[i]; ih = ch[i + 1] - ch[i];
                id0 = dv[i]; id1 = dv[i + 1];
            }
        }

        // ---- spline + log-det ----
        float delta = ih / iw;
        float th  = (xin - icw) / iw;
        float tt  = th * (1.f - th);
        float numer = ih * (delta * th * th + id0 * tt);
        float denom = delta + (id0 + id1 - 2.f * delta) * tt;
        float outv  = ich + numer / denom;
        float omt   = 1.f - th;
        float dnum  = delta * delta * (id1 * th * th + 2.f * delta * tt + id0 * omt * omt);
        float lad   = __logf(dnum) - 2.f * __logf(denom);

        tgt[row * 32 + j] = inside ? outv : xv;
        ldsum += inside ? lad : 0.f;
    }
    atomicAdd(&ldacc[row], ldsum);     // ds_add_f32
}

// One coupling: MLP(src) -> params -> RQS transform of tgt (both 32x32 in LDS)
__device__ void coupling(char* sm, const float* srcbuf, float* tgtbuf,
                         const __bf16* wt0, const float* b0,
                         const __bf16* wt1, const float* b1,
                         const __bf16* wt2, const float* b2, int tid) {
    __bf16* AH = (__bf16*)(sm + OFF_AH);
    __bf16* AL = (__bf16*)(sm + OFF_AL);
    __bf16* HH = (__bf16*)(sm + OFF_HH);
    __bf16* HL = (__bf16*)(sm + OFF_HL);
    __bf16* WB = (__bf16*)(sm + OFF_WB);
    float*  OB = (float*)(sm + OFF_OUT);
    float*  LD = (float*)(sm + OFF_LD);

    // split src (f32) into bf16 hi/lo activation buffers (stride 32)
    for (int idx = tid; idx < ROWS * 32; idx += THREADS) {
        float v = srcbuf[idx];
        __bf16 h = (__bf16)v;
        AH[idx] = h;
        AL[idx] = (__bf16)(v - (float)h);
    }
    // (gemm_layer's leading __syncthreads covers the fill above)
    gemm_layer<32,  HID,  32,  4, true >(AH, AL, wt0, b0, WB, HH, HL, nullptr, tid);
    gemm_layer<HID, HID,  HID, 4, true >(HH, HL, wt1, b1, WB, AH, AL, nullptr, tid);
    gemm_layer<HID, OUTD, HID, 12, false>(AH, AL, wt2, b2, WB, nullptr, nullptr, OB, tid);
    rqs_block(OB, tgtbuf, LD, tid);
    __syncthreads();
}

// Prep: W [KTOT][NOUT] f32 -> slab-major transposed bf16 hi/lo:
//   dst + s*(NOUT*64) : [NOUT*32 hi][NOUT*32 lo], each [out][k%32]
__global__ void __launch_bounds__(THREADS)
prep_weights_kernel(const float* __restrict__ W, __bf16* __restrict__ dst,
                    int ktot, int nout) {
    int idx = blockIdx.x * THREADS + threadIdx.x;
    if (idx >= ktot * nout) return;
    int k = idx / nout;
    int o = idx - k * nout;
    int s = k >> 5, kk = k & 31;
    float v = W[idx];
    __bf16 h = (__bf16)v;
    float  r = v - (float)h;
    size_t slab = (size_t)s * (nout * 64);
    dst[slab + o * 32 + kk] = h;
    dst[slab + nout * 32 + o * 32 + kk] = (__bf16)r;
}

__global__ void __launch_bounds__(THREADS)
nsf_fused_kernel(const float* __restrict__ x,
                 const __bf16* wt0a, const __bf16* wt1a, const __bf16* wt2a,
                 const __bf16* wt0b, const __bf16* wt1b, const __bf16* wt2b,
                 const float* b0a, const float* b1a, const float* b2a,
                 const float* b0b, const float* b1b, const float* b2b,
                 float* __restrict__ outx, float* __restrict__ outld) {
    extern __shared__ char sm[];
    float* XL = (float*)(sm + OFF_XLOW);
    float* XU = (float*)(sm + OFF_XUP);
    float* LD = (float*)(sm + OFF_LD);

    const int tid = threadIdx.x;
    const size_t r0 = (size_t)blockIdx.x * ROWS;

    for (int idx = tid; idx < ROWS * 64; idx += THREADS) {
        int r = idx >> 6, cc = idx & 63;
        float v = x[(r0 + r) * 64 + cc];
        if (cc < 32) XL[r * 32 + cc] = v;
        else         XU[r * 32 + (cc - 32)] = v;
    }
    if (tid < ROWS) LD[tid] = 0.f;

    coupling(sm, XL, XU, wt0a, b0a, wt1a, b1a, wt2a, b2a, tid);   // transform upper
    coupling(sm, XU, XL, wt0b, b0b, wt1b, b1b, wt2b, b2b, tid);   // transform lower

    for (int idx = tid; idx < ROWS * 64; idx += THREADS) {
        int r = idx >> 6, cc = idx & 63;
        float v = (cc < 32) ? XL[r * 32 + cc] : XU[r * 32 + (cc - 32)];
        outx[(r0 + r) * 64 + cc] = v;
    }
    for (int r = tid; r < ROWS; r += THREADS) outld[r0 + r] = LD[r];
}

extern "C" void kernel_launch(void* const* d_in, const int* in_sizes, int n_in,
                              void* d_out, int out_size, void* d_ws, size_t ws_size,
                              hipStream_t stream) {
    const float* x   = (const float*)d_in[0];
    const float* w0a = (const float*)d_in[1];
    const float* b0a = (const float*)d_in[2];
    const float* w1a = (const float*)d_in[3];
    const float* b1a = (const float*)d_in[4];
    const float* w2a = (const float*)d_in[5];
    const float* b2a = (const float*)d_in[6];
    const float* w0b = (const float*)d_in[7];
    const float* b0b = (const float*)d_in[8];
    const float* w1b = (const float*)d_in[9];
    const float* b1b = (const float*)d_in[10];
    const float* w2b = (const float*)d_in[11];
    const float* b2b = (const float*)d_in[12];

    // Workspace layout (bf16 elements). Per MLP: w0t 16384, w1t 131072,
    // w2t 376832 -> 524288 elems (1MB). Total 2MB.
    __bf16* wt   = (__bf16*)d_ws;
    __bf16* wt0a = wt;
    __bf16* wt1a = wt + 16384;
    __bf16* wt2a = wt + 147456;
    __bf16* wt0b = wt + 524288;
    __bf16* wt1b = wt + 540672;
    __bf16* wt2b = wt + 671744;

    prep_weights_kernel<<<(32  * 256 + THREADS - 1) / THREADS, THREADS, 0, stream>>>(w0a, wt0a, 32,  256);
    prep_weights_kernel<<<(256 * 256 + THREADS - 1) / THREADS, THREADS, 0, stream>>>(w1a, wt1a, 256, 256);
    prep_weights_kernel<<<(256 * 736 + THREADS - 1) / THREADS, THREADS, 0, stream>>>(w2a, wt2a, 256, 736);
    prep_weights_kernel<<<(32  * 256 + THREADS - 1) / THREADS, THREADS, 0, stream>>>(w0b, wt0b, 32,  256);
    prep_weights_kernel<<<(256 * 256 + THREADS - 1) / THREADS, THREADS, 0, stream>>>(w1b, wt1b, 256, 256);
    prep_weights_kernel<<<(256 * 736 + THREADS - 1) / THREADS, THREADS, 0, stream>>>(w2b, wt2b, 256, 736);

    const int n_rows = in_sizes[0] / 64;                 // N = 65536
    float* outx  = (float*)d_out;                        // N x 64
    float* outld = (float*)d_out + (size_t)n_rows * 64;  // N

    dim3 grid(n_rows / ROWS);
    nsf_fused_kernel<<<grid, THREADS, SMEM_BYTES, stream>>>(
        x, wt0a, wt1a, wt2a, wt0b, wt1b, wt2b,
        b0a, b1a, b2a, b0b, b1b, b2b, outx, outld);
}
